// LstmLayer_54546084659584
// MI455X (gfx1250) — matrix operand order repaired and checked by reading, not compiled, not run on hardware
//
#include <hip/hip_runtime.h>

#define N_STEPS 24
#define IN_LEN 512
#define OUT_LEN 256
#define H_LEN 1024
#define C_LEN 1024
#define BATCH 128
#define IH (IN_LEN + H_LEN) /* 1536 */

#define BPAD 40  /* padded bf16 row length (32 data + 8 pad) -> 80B row stride */

typedef __attribute__((ext_vector_type(16))) __bf16 v16bf;
typedef __attribute__((ext_vector_type(8)))  __bf16 v8bf;
typedef __attribute__((ext_vector_type(2)))  __bf16 v2bf;
typedef __attribute__((ext_vector_type(8)))  float  v8f;
typedef __attribute__((ext_vector_type(4)))  float  f4v;
typedef __attribute__((ext_vector_type(2)))  float  f2v;

union ABf { v16bf v; v8bf h[2]; };

// Hardware tanh on gfx1250 (V_TANH_F32, single TRANS op) with libm fallback.
#if __has_builtin(__builtin_amdgcn_tanhf)
__device__ __forceinline__ float tanh_fast(float x) { return __builtin_amdgcn_tanhf(x); }
#else
__device__ __forceinline__ float tanh_fast(float x) { return tanhf(x); }
#endif

// sigmoid(x) = 0.5 + 0.5*tanh(x/2): 3 VALU ops with hardware tanh.
__device__ __forceinline__ float sigmoid_fast(float x) {
    return 0.5f + 0.5f * tanh_fast(0.5f * x);
}

__device__ __forceinline__ v8bf cvt8_(f4v a, f4v b) {
    v8bf r;
    r[0] = (__bf16)a.x; r[1] = (__bf16)a.y; r[2] = (__bf16)a.z; r[3] = (__bf16)a.w;
    r[4] = (__bf16)b.x; r[5] = (__bf16)b.y; r[6] = (__bf16)b.z; r[7] = (__bf16)b.w;
    return r;
}

// Broadcast c0 -> c[128,1024] (fp32) and h0 -> bf16 h-region of xh buffer 0.
__global__ void init_state(const float* __restrict__ c0, const float* __restrict__ h0,
                           float* __restrict__ c, __bf16* __restrict__ xh0) {
    int idx = blockIdx.x * blockDim.x + threadIdx.x;   // 0 .. 128*1024/8-1
    int bb = idx >> 7;
    int j  = (idx & 127) * 8;
    f4v c0a = *(const f4v*)(c0 + j);
    f4v c0b = *(const f4v*)(c0 + j + 4);
    f4v h0a = *(const f4v*)(h0 + j);
    f4v h0b = *(const f4v*)(h0 + j + 4);
    f4v* cd = (f4v*)(c + (size_t)bb * C_LEN + j);
    cd[0] = c0a; cd[1] = c0b;
    *(v8bf*)(xh0 + (size_t)bb * IH + IN_LEN + j) = cvt8_(h0a, h0b);
}

// Convert x[:, t, :] fp32 -> bf16 into xh (cols 0..511). 8 elems per thread.
__global__ void stage_x(const float* __restrict__ x, __bf16* __restrict__ xh, int t) {
    int idx = blockIdx.x * blockDim.x + threadIdx.x;   // 0 .. 128*512/8-1
    int bb = idx >> 6;
    int i  = (idx & 63) * 8;
    const float* src = x + (size_t)bb * N_STEPS * IN_LEN + (size_t)t * IN_LEN + i;
    f4v a = *(const f4v*)(src);
    f4v b = *(const f4v*)(src + 4);
    *(v8bf*)(xh + (size_t)bb * IH + i) = cvt8_(a, b);
}

// One K-step for the 4-gate GEMM: 8 ds_load_b128 up front, then 4 WMMAs.
#define GATES_STEP(LDSBUF, J) do {                                            \
    const __bf16* ap_ = aptr + (J) * 32;                                      \
    ABf af_;                                                                  \
    af_.h[0] = *(const v8bf*)(ap_);                                           \
    af_.h[1] = *(const v8bf*)(ap_ + 16);                                      \
    const __bf16* bp0_ = &LDSBUF[0][lmod][lhalf * 8];                         \
    const __bf16* bp1_ = &LDSBUF[1][lmod][lhalf * 8];                         \
    const __bf16* bp2_ = &LDSBUF[2][lmod][lhalf * 8];                         \
    const __bf16* bp3_ = &LDSBUF[3][lmod][lhalf * 8];                         \
    ABf b0_, b1_, b2_, b3_;                                                   \
    b0_.h[0] = *(const v8bf*)(bp0_); b0_.h[1] = *(const v8bf*)(bp0_ + 16);    \
    b1_.h[0] = *(const v8bf*)(bp1_); b1_.h[1] = *(const v8bf*)(bp1_ + 16);    \
    b2_.h[0] = *(const v8bf*)(bp2_); b2_.h[1] = *(const v8bf*)(bp2_ + 16);    \
    b3_.h[0] = *(const v8bf*)(bp3_); b3_.h[1] = *(const v8bf*)(bp3_ + 16);    \
    acc0 = __builtin_amdgcn_wmma_f32_16x16x32_bf16(                           \
        false, af_.v, false, b0_.v, (short)0, acc0, false, false);            \
    acc1 = __builtin_amdgcn_wmma_f32_16x16x32_bf16(                           \
        false, af_.v, false, b1_.v, (short)0, acc1, false, false);            \
    acc2 = __builtin_amdgcn_wmma_f32_16x16x32_bf16(                           \
        false, af_.v, false, b2_.v, (short)0, acc2, false, false);            \
    acc3 = __builtin_amdgcn_wmma_f32_16x16x32_bf16(                           \
        false, af_.v, false, b3_.v, (short)0, acc3, false, false);            \
} while (0)

// Weight slabs are single-use -> non-temporal loads keep L2 for reused xh.
#define GATES_PREFETCH(Q0, Q1, J) do {                                        \
    const f4v* wp_ = (const f4v*)(wrow + (J) * 32);                           \
    (Q0) = __builtin_nontemporal_load(wp_);                                   \
    (Q1) = __builtin_nontemporal_load(wp_ + 1);                               \
} while (0)

#define GATES_STORE_SLAB(LDSBUF, Q0, Q1)                                      \
    *(v8bf*)&LDSBUF[lg][ln][lk] = cvt8_((Q0), (Q1))

// Fused 4-gate GEMM + LSTM cell update.
// Block: 256 threads = 8 wave32. Wave w -> M tile rows [16w,16w+16).
// Block covers one 16-wide N tile for ALL four gates.
// fp32 weights stream from HBM (non-temporal), converted to bf16 ONCE by the
// cooperative loader into two statically-addressed LDS buffers (manual
// 2x-unrolled software pipeline).
__launch_bounds__(256)
__global__ void lstm_gates(const __bf16* __restrict__ xh,
                           const float* __restrict__ W,  const float* __restrict__ Wi,
                           const float* __restrict__ Wf, const float* __restrict__ Wo,
                           const float* __restrict__ b,  const float* __restrict__ bi,
                           const float* __restrict__ bfb, const float* __restrict__ bo,
                           float* __restrict__ c, __bf16* __restrict__ xh_next, int t)
{
    __shared__ __bf16 lds_b0[4][16][BPAD];
    __shared__ __bf16 lds_b1[4][16][BPAD];

    const int tid   = threadIdx.x;
    const int lane  = tid & 31;
    const int wave  = tid >> 5;              // 0..7 -> M tile
    const int nbase = blockIdx.x * 16;       // N tile base
    const int lhalf = lane >> 4;             // 0/1
    const int lmod  = lane & 15;

    const size_t wofs = (size_t)t * C_LEN * IH;
    // Cooperative slab loader: 64 threads per gate, 16 n-rows x 4 k-chunks of 8.
    const int lg = tid >> 6;                 // gate
    const int lr = tid & 63;
    const int ln = lr >> 2;                  // n within tile
    const int lk = (lr & 3) * 8;             // k offset (0/8/16/24)
    const float* Wg = (lg == 0) ? W : (lg == 1) ? Wi : (lg == 2) ? Wf : Wo;
    const float* wrow = Wg + wofs + (size_t)(nbase + ln) * IH + lk;

    // A-fragment base: lane -> row (l&15), k halves at (l>>4)*8 and +16.
    const int mrow = wave * 16 + lmod;
    const __bf16* aptr = xh + (size_t)mrow * IH + lhalf * 8;

    v8f acc0 = {}, acc1 = {}, acc2 = {}, acc3 = {};
    const int NK = IH / 32;   // 48 K-steps (even)

    // Preload slab 0 into lds_b0.
    {
        f4v p0, p1;
        GATES_PREFETCH(p0, p1, 0);
        GATES_STORE_SLAB(lds_b0, p0, p1);
    }
    __syncthreads();

    // Steady state: two slabs per trip, unconditional prefetch.
    for (int jj = 0; jj < NK - 2; jj += 2) {
        f4v q0, q1;
        GATES_PREFETCH(q0, q1, jj + 1);
        GATES_STEP(lds_b0, jj);
        GATES_STORE_SLAB(lds_b1, q0, q1);
        __syncthreads();

        f4v r0, r1;
        GATES_PREFETCH(r0, r1, jj + 2);
        GATES_STEP(lds_b1, jj + 1);
        GATES_STORE_SLAB(lds_b0, r0, r1);
        __syncthreads();
    }
    // Drain: lds_b0 holds slab NK-2.
    {
        f4v q0, q1;
        GATES_PREFETCH(q0, q1, NK - 1);
        GATES_STEP(lds_b0, NK - 2);
        GATES_STORE_SLAB(lds_b1, q0, q1);
        __syncthreads();
        GATES_STEP(lds_b1, NK - 1);
    }

    // Epilogue: fused LSTM cell update with hardware tanh.
    // C/D layout: VGPR i, lanes 0-15 -> M=i, lanes 16-31 -> M=8+i; N = lane&15.
    const int nn = nbase + lmod;
    const float bz  = b  [(size_t)t * C_LEN + nn];
    const float bzi = bi [(size_t)t * C_LEN + nn];
    const float bzf = bfb[(size_t)t * C_LEN + nn];
    const float bzo = bo [(size_t)t * C_LEN + nn];

    #pragma unroll
    for (int i = 0; i < 8; ++i) {
        const int m = wave * 16 + i + 8 * lhalf;
        float z  = tanh_fast(acc0[i] + bz);
        float zi = sigmoid_fast(acc1[i] + bzi);
        float zf = sigmoid_fast(acc2[i] + bzf);
        float zo = sigmoid_fast(acc3[i] + bzo);
        const size_t ci = (size_t)m * C_LEN + nn;
        float cn = zf * c[ci] + zi * z;
        c[ci] = cn;
        float hn = zo * tanh_fast(cn);
        xh_next[(size_t)m * IH + IN_LEN + nn] = (__bf16)hn;
    }
}

// One K-step for the output GEMM.
#define OUT_STEP(LDSBUF, J) do {                                              \
    const __bf16* ap_ = aptr + (J) * 32;                                      \
    ABf af_;                                                                  \
    af_.h[0] = *(const v8bf*)(ap_);                                           \
    af_.h[1] = *(const v8bf*)(ap_ + 16);                                      \
    const __bf16* bp_ = &LDSBUF[lmod][lhalf * 8];                             \
    ABf bf_;                                                                  \
    bf_.h[0] = *(const v8bf*)(bp_);                                           \
    bf_.h[1] = *(const v8bf*)(bp_ + 16);                                      \
    acc0 = __builtin_amdgcn_wmma_f32_16x16x32_bf16(                           \
        false, af_.v, false, bf_.v, (short)0, acc0, false, false);            \
} while (0)

#define OUT_STORE_SLAB(LDSBUF, Q) do {                                        \
    v2bf r_; r_[0] = (__bf16)(Q).x; r_[1] = (__bf16)(Q).y;                    \
    *(v2bf*)&LDSBUF[ln][lk] = r_;                                             \
} while (0)

// y = sigmoid(h @ Wout^T + bout)  : [128,1024] x [1024->256]
// Block: 256 threads = 8 wave32; wave w -> M tile, block -> one 16-wide N tile.
__launch_bounds__(256)
__global__ void lstm_out(const __bf16* __restrict__ xh_next,
                         const float* __restrict__ Wout,
                         const float* __restrict__ bout,
                         float* __restrict__ out, int t)
{
    __shared__ __bf16 lds_c0[16][BPAD];
    __shared__ __bf16 lds_c1[16][BPAD];

    const int tid   = threadIdx.x;
    const int lane  = tid & 31;
    const int wave  = tid >> 5;              // M tile
    const int nbase = blockIdx.x * 16;       // N tile base
    const int lhalf = lane >> 4;
    const int lmod  = lane & 15;

    // Loader: 16 threads per n-row, 2 floats each (covers 16n x 32k slab).
    const int ln = tid >> 4;
    const int lk = (tid & 15) * 2;
    const float* wrow = Wout + (size_t)t * OUT_LEN * H_LEN
                             + (size_t)(nbase + ln) * H_LEN + lk;

    const int m = wave * 16 + lmod;
    const __bf16* aptr = xh_next + (size_t)m * IH + IN_LEN + lhalf * 8;

    v8f acc0 = {};
    const int NK = H_LEN / 32;   // 32 K-steps (even)

    {
        f2v p = __builtin_nontemporal_load((const f2v*)(wrow));
        OUT_STORE_SLAB(lds_c0, p);
    }
    __syncthreads();

    for (int jj = 0; jj < NK - 2; jj += 2) {
        f2v q = __builtin_nontemporal_load((const f2v*)(wrow + (jj + 1) * 32));
        OUT_STEP(lds_c0, jj);
        OUT_STORE_SLAB(lds_c1, q);
        __syncthreads();

        f2v r = __builtin_nontemporal_load((const f2v*)(wrow + (jj + 2) * 32));
        OUT_STEP(lds_c1, jj + 1);
        OUT_STORE_SLAB(lds_c0, r);
        __syncthreads();
    }
    {
        f2v q = __builtin_nontemporal_load((const f2v*)(wrow + (NK - 1) * 32));
        OUT_STEP(lds_c0, NK - 2);
        OUT_STORE_SLAB(lds_c1, q);
        __syncthreads();
        OUT_STEP(lds_c1, NK - 1);
    }

    const int nn = nbase + lmod;
    const float bb = bout[(size_t)t * OUT_LEN + nn];
    #pragma unroll
    for (int i = 0; i < 8; ++i) {
        const int mm = wave * 16 + i + 8 * lhalf;
        out[(size_t)mm * (N_STEPS * OUT_LEN) + (size_t)t * OUT_LEN + nn] =
            sigmoid_fast(acc0[i] + bb);
    }
}

extern "C" void kernel_launch(void* const* d_in, const int* in_sizes, int n_in,
                              void* d_out, int out_size, void* d_ws, size_t ws_size,
                              hipStream_t stream) {
    const float* x    = (const float*)d_in[0];
    const float* W    = (const float*)d_in[1];
    const float* Wi   = (const float*)d_in[2];
    const float* Wf   = (const float*)d_in[3];
    const float* Wo   = (const float*)d_in[4];
    const float* Wout = (const float*)d_in[5];
    const float* b    = (const float*)d_in[6];
    const float* bi   = (const float*)d_in[7];
    const float* bf   = (const float*)d_in[8];
    const float* bo   = (const float*)d_in[9];
    const float* bout = (const float*)d_in[10];
    const float* c0   = (const float*)d_in[11];
    const float* h0   = (const float*)d_in[12];
    float* out = (float*)d_out;

    // Workspace: two ping-pong bf16 xh buffers + fp32 cell state (~1.31 MB).
    __bf16* xh0 = (__bf16*)d_ws;
    __bf16* xh1 = xh0 + (size_t)BATCH * IH;
    float*  c   = (float*)((char*)d_ws + 2u * (size_t)BATCH * IH * sizeof(__bf16));
    __bf16* xhbuf[2] = { xh0, xh1 };

    init_state<<<(BATCH * C_LEN / 8) / 256, 256, 0, stream>>>(c0, h0, c, xh0);

    for (int t = 0; t < N_STEPS; ++t) {
        __bf16* cur = xhbuf[t & 1];
        __bf16* nxt = xhbuf[(t & 1) ^ 1];
        stage_x<<<(BATCH * IN_LEN / 8) / 256, 256, 0, stream>>>(x, cur, t);
        lstm_gates<<<C_LEN / 16, 256, 0, stream>>>(cur, W, Wi, Wf, Wo,
                                                   b, bi, bf, bo, c, nxt, t);
        lstm_out<<<OUT_LEN / 16, 256, 0, stream>>>(nxt, Wout, bout, out, t);
    }
}
